// DecoderAttention_86346022518972
// MI455X (gfx1250) — matrix operand order repaired and checked by reading, not compiled, not run on hardware
//
#include <hip/hip_runtime.h>
#include <math.h>

// ---------------- problem constants ----------------
#define HEADS 8
#define KD    64
#define VD    64
#define KNN   128
#define DIM   512
#define FCD   2048
#define NB    4
#define SEQ   1024
#define MROWS (NB * SEQ)   // 4096
#define EPS   1e-5f

typedef __attribute__((ext_vector_type(16))) _Float16 v16h;
typedef __attribute__((ext_vector_type(8)))  float    v8f;

// =====================================================================
// Generic batched WMMA GEMM:  C = alpha * (A @ B) + bias, optional ReLU
//   A element (m,k)  : A[z*a_bs + m*lda_m + k]            (k contiguous)
//   B element (k,n)  : B[z*b_bs + k*ldb_k + n*ldb_n]
//       BMODE==0 : ldb_k==1  (k-contiguous source, e.g. K^T for scores)
//       BMODE==1 : ldb_n==1  (n-contiguous source, e.g. weights)
//   C element (m,n)  : C[(z/cdiv)*c_bso + (z%cdiv)*c_bsi + m*ldc_m + n]
//   bias (optional)  : bias[z*bias_bs + n]
// Block = 128 threads (4 waves), tile 64x64, BK=32.
// Double-buffered f32->f16 LDS staging; B tile stored transposed so both
// A and B fragments are contiguous 16B runs -> ds_load_b128.
// =====================================================================
#define BM 64
#define BN 64
#define BK 32

template<int BMODE>
__global__ __launch_bounds__(128)
void gemm_wmma_f16(const float* __restrict__ A, const float* __restrict__ B,
                   const float* __restrict__ bias, float* __restrict__ C,
                   int M, int Nc, int K,
                   long lda_m, long a_bs,
                   long ldb_k, long ldb_n, long b_bs,
                   long ldc_m, long c_bso, long c_bsi, int cdiv,
                   long bias_bs, float alpha, int relu)
{
    __shared__ _Float16 As [2][BM][BK + 8];   // [buf][m][k]
    __shared__ _Float16 Bst[2][BN][BK + 8];   // [buf][n][k]  (transposed tile)

    const int z = blockIdx.z;
    A += (long)z * a_bs;
    B += (long)z * b_bs;
    const long c_off = (long)(z / cdiv) * c_bso + (long)(z % cdiv) * c_bsi;

    const int m0   = blockIdx.y * BM;
    const int n0   = blockIdx.x * BN;
    const int tid  = threadIdx.x;
    const int wave = tid >> 5;
    const int lane = tid & 31;
    const int wm   = (wave >> 1) * 32;   // wave row offset in tile
    const int wn   = (wave & 1) * 32;    // wave col offset in tile

    v8f acc[2][2];
    for (int i = 0; i < 2; ++i)
        for (int j = 0; j < 2; ++j)
            acc[i][j] = (v8f){};

    const int mrow = lane & 15;
    const int kh   = lane >> 4;          // which K-half this lane owns

    auto stageA = [&](int buf, int k0) {
        #pragma unroll
        for (int e = tid; e < BM * (BK / 4); e += 128) {   // BK/4 == 8
            const int r  = e >> 3;
            const int c4 = (e & 7) * 4;
            const float* src = &A[(long)(m0 + r) * lda_m + (k0 + c4)];
            __builtin_prefetch(src + BK, 0, 3);
            const float4 f = *(const float4*)src;
            _Float16* dst = &As[buf][r][c4];
            dst[0] = (_Float16)f.x; dst[1] = (_Float16)f.y;
            dst[2] = (_Float16)f.z; dst[3] = (_Float16)f.w;
        }
    };
    auto stageB = [&](int buf, int k0) {
        if (BMODE == 0) {
            // k-contiguous source: float4 along k, contiguous LDS store
            #pragma unroll
            for (int e = tid; e < BN * (BK / 4); e += 128) {
                const int n  = e >> 3;
                const int c4 = (e & 7) * 4;
                const float* src = &B[(long)(k0 + c4) + (long)(n0 + n) * ldb_n];
                __builtin_prefetch(src + BK, 0, 3);
                const float4 f = *(const float4*)src;
                _Float16* dst = &Bst[buf][n][c4];
                dst[0] = (_Float16)f.x; dst[1] = (_Float16)f.y;
                dst[2] = (_Float16)f.z; dst[3] = (_Float16)f.w;
            }
        } else {
            // n-contiguous source: float4 along n, scatter into 4 LDS rows
            #pragma unroll
            for (int e = tid; e < BK * (BN / 4); e += 128) {  // BN/4 == 16
                const int r  = e >> 4;          // k index
                const int c4 = (e & 15) * 4;    // n index
                const float* src = &B[(long)(k0 + r) * ldb_k + (n0 + c4)];
                __builtin_prefetch(src + BK * ldb_k, 0, 3);
                const float4 f = *(const float4*)src;
                Bst[buf][c4 + 0][r] = (_Float16)f.x;
                Bst[buf][c4 + 1][r] = (_Float16)f.y;
                Bst[buf][c4 + 2][r] = (_Float16)f.z;
                Bst[buf][c4 + 3][r] = (_Float16)f.w;
            }
        }
    };

    // prologue: stage first tile
    stageA(0, 0);
    stageB(0, 0);
    __syncthreads();

    for (int k0 = 0; k0 < K; k0 += BK) {
        const int p = (k0 / BK) & 1;
        // stage next tile into the other buffer while computing this one
        if (k0 + BK < K) {
            stageA(1 - p, k0 + BK);
            stageB(1 - p, k0 + BK);
        }

        v16h afrag[2], bfrag[2];
        #pragma unroll
        for (int i = 0; i < 2; ++i) {
            const int mb = wm + i * 16 + mrow;
            #pragma unroll
            for (int t = 0; t < 8; ++t) {
                afrag[i][t]     = As[p][mb][kh * 8 + t];        // K 0..7   / 8..15
                afrag[i][t + 8] = As[p][mb][16 + kh * 8 + t];   // K 16..23 / 24..31
            }
        }
        #pragma unroll
        for (int j = 0; j < 2; ++j) {
            const int nb = wn + j * 16 + mrow;
            #pragma unroll
            for (int t = 0; t < 8; ++t) {
                bfrag[j][t]     = Bst[p][nb][kh * 8 + t];       // K 0..7   / 8..15
                bfrag[j][t + 8] = Bst[p][nb][16 + kh * 8 + t];  // K 16..23 / 24..31
            }
        }

        #pragma unroll
        for (int i = 0; i < 2; ++i)
            #pragma unroll
            for (int j = 0; j < 2; ++j)
                acc[i][j] = __builtin_amdgcn_wmma_f32_16x16x32_f16(
                    false, afrag[i], false, bfrag[j],
                    (short)0, acc[i][j], false, false);
        __syncthreads();
    }

    // epilogue: C/D layout -> row = v + 8*(lane>>4), col = lane&15
    const int nlane = lane & 15;
    const int rhalf = (lane >> 4) * 8;
    for (int i = 0; i < 2; ++i) {
        for (int j = 0; j < 2; ++j) {
            const int col = n0 + wn + j * 16 + nlane;
            const float bv = bias ? bias[(long)z * bias_bs + col] : 0.f;
            #pragma unroll
            for (int v = 0; v < 8; ++v) {
                const int row = m0 + wm + i * 16 + rhalf + v;
                float val = alpha * acc[i][j][v] + bv;
                if (relu) val = fmaxf(val, 0.f);
                C[c_off + (long)row * ldc_m + col] = val;
            }
        }
    }
}

// =====================================================================
// Row softmax (for the graph matrices), one 256-thread block per row
// =====================================================================
__global__ __launch_bounds__(256)
void row_softmax(const float* __restrict__ in, float* __restrict__ out, int cols)
{
    __shared__ float red[256];
    const int row = blockIdx.x;
    const int tid = threadIdx.x;
    const float* x = in + (long)row * cols;

    float m = -INFINITY;
    for (int c = tid; c < cols; c += 256) m = fmaxf(m, x[c]);
    red[tid] = m; __syncthreads();
    for (int s = 128; s > 0; s >>= 1) {
        if (tid < s) red[tid] = fmaxf(red[tid], red[tid + s]);
        __syncthreads();
    }
    m = red[0]; __syncthreads();

    float sum = 0.f;
    for (int c = tid; c < cols; c += 256) sum += __expf(x[c] - m);
    red[tid] = sum; __syncthreads();
    for (int s = 128; s > 0; s >>= 1) {
        if (tid < s) red[tid] += red[tid + s];
        __syncthreads();
    }
    const float inv = 1.f / red[0];

    for (int c = tid; c < cols; c += 256)
        out[(long)row * cols + c] = __expf(x[c] - m) * inv;
}

// =====================================================================
// Top-KNN (exact 128th-largest via 4-pass radix select on ordered bits)
// + softmax over kept + blend with softmaxed graph. In-place on att row.
// grid = (SEQ, H*N), block = 256. causal => row length q+1.
// =====================================================================
__device__ __forceinline__ unsigned ordf(float x)
{
    unsigned u = __float_as_uint(x);
    return (u & 0x80000000u) ? ~u : (u | 0x80000000u);
}

__global__ __launch_bounds__(256)
void topk_softmax_blend(float* __restrict__ att, const float* __restrict__ graph,
                        int causal, float gw)
{
    __shared__ unsigned hist[256];
    __shared__ float    fred[256];
    __shared__ unsigned spfx;
    __shared__ int      sk;

    const int q = blockIdx.x;
    const int z = blockIdx.y;
    const int tid = threadIdx.x;
    float* x = att + ((long)z * SEQ + q) * SEQ;
    const int L = causal ? (q + 1) : SEQ;

    unsigned thresh = 0;   // keep-all: ordf(v) >= 0 always true
    if (L > KNN) {
        unsigned prefix = 0;
        if (tid == 0) sk = KNN;
        __syncthreads();
        for (int pass = 3; pass >= 0; --pass) {
            const int shift = pass * 8;
            hist[tid] = 0;
            __syncthreads();
            const unsigned maskhi = (pass == 3) ? 0u : (0xFFFFFFFFu << (shift + 8));
            for (int s = tid; s < L; s += 256) {
                unsigned u = ordf(x[s]);
                if ((u & maskhi) == prefix)
                    atomicAdd(&hist[(u >> shift) & 255], 1u);
            }
            __syncthreads();
            if (tid == 0) {
                int kk = sk;
                unsigned cum = 0; int b = 255;
                for (; b >= 0; --b) { cum += hist[b]; if ((int)cum >= kk) break; }
                sk = kk - (int)(cum - hist[b]);
                spfx = prefix | (((unsigned)b) << shift);
            }
            __syncthreads();
            prefix = spfx;
            __syncthreads();
        }
        thresh = prefix;   // exact bits of the 128th-largest value
    }

    // row max over kept (== overall max of first L)
    float m = -INFINITY;
    for (int s = tid; s < L; s += 256) {
        float v = x[s];
        if (ordf(v) >= thresh) m = fmaxf(m, v);
    }
    fred[tid] = m; __syncthreads();
    for (int s = 128; s > 0; s >>= 1) {
        if (tid < s) fred[tid] = fmaxf(fred[tid], fred[tid + s]);
        __syncthreads();
    }
    m = fred[0]; __syncthreads();

    float sum = 0.f;
    for (int s = tid; s < L; s += 256) {
        float v = x[s];
        if (ordf(v) >= thresh) sum += __expf(v - m);
    }
    fred[tid] = sum; __syncthreads();
    for (int s = 128; s > 0; s >>= 1) {
        if (tid < s) fred[tid] += fred[tid + s];
        __syncthreads();
    }
    const float inv = 1.f / fred[0];

    const float* g = graph + (long)q * SEQ;
    for (int s = tid; s < SEQ; s += 256) {
        float p = 0.f;
        if (s < L) {
            float v = x[s];
            if (ordf(v) >= thresh) p = __expf(v - m) * inv;
        }
        x[s] = gw * g[s] + (1.f - gw) * p;
    }
}

// =====================================================================
// out = LayerNorm(a + b), rows of width DIM=512, one block (256thr) per row
// =====================================================================
__global__ __launch_bounds__(256)
void add_layernorm(const float* __restrict__ a, const float* __restrict__ b,
                   float* __restrict__ out)
{
    __shared__ float red[256];
    const int row = blockIdx.x;
    const int tid = threadIdx.x;
    const long base = (long)row * DIM;

    float x0 = a[base + tid]       + b[base + tid];
    float x1 = a[base + tid + 256] + b[base + tid + 256];

    red[tid] = x0 + x1; __syncthreads();
    for (int s = 128; s > 0; s >>= 1) {
        if (tid < s) red[tid] += red[tid + s];
        __syncthreads();
    }
    const float mean = red[0] / (float)DIM;
    __syncthreads();

    float d0 = x0 - mean, d1 = x1 - mean;
    red[tid] = d0 * d0 + d1 * d1; __syncthreads();
    for (int s = 128; s > 0; s >>= 1) {
        if (tid < s) red[tid] += red[tid + s];
        __syncthreads();
    }
    const float rstd = rsqrtf(red[0] / (float)DIM + EPS);

    out[base + tid]       = d0 * rstd;
    out[base + tid + 256] = d1 * rstd;
}

// =====================================================================
// Host-side orchestration
// =====================================================================
static void launch_gemm(hipStream_t st, int bmode,
                        const float* A, const float* B, const float* bias, float* C,
                        int M, int Nc, int K,
                        long lda_m, long a_bs,
                        long ldb_k, long ldb_n, long b_bs,
                        long ldc_m, long c_bso, long c_bsi, int cdiv,
                        long bias_bs, float alpha, int relu, int batch)
{
    dim3 grid(Nc / BN, M / BM, batch);
    if (bmode == 0)
        gemm_wmma_f16<0><<<grid, 128, 0, st>>>(A, B, bias, C, M, Nc, K,
                                               lda_m, a_bs, ldb_k, ldb_n, b_bs,
                                               ldc_m, c_bso, c_bsi, cdiv,
                                               bias_bs, alpha, relu);
    else
        gemm_wmma_f16<1><<<grid, 128, 0, st>>>(A, B, bias, C, M, Nc, K,
                                               lda_m, a_bs, ldb_k, ldb_n, b_bs,
                                               ldc_m, c_bso, c_bsi, cdiv,
                                               bias_bs, alpha, relu);
}

extern "C" void kernel_launch(void* const* d_in, const int* in_sizes, int n_in,
                              void* d_out, int out_size, void* d_ws, size_t ws_size,
                              hipStream_t stream)
{
    const float* z_in   = (const float*)d_in[0];
    const float* y_in   = (const float*)d_in[1];
    const float* g_dec  = (const float*)d_in[2];
    const float* g_enc  = (const float*)d_in[3];
    const float* dec_Wk = (const float*)d_in[4];
    const float* dec_bk = (const float*)d_in[5];
    const float* dec_Wv = (const float*)d_in[6];
    const float* dec_bv = (const float*)d_in[7];
    const float* dec_Wo = (const float*)d_in[8];
    const float* dec_bo = (const float*)d_in[9];
    const float* enc_Wk = (const float*)d_in[10];
    const float* enc_bk = (const float*)d_in[11];
    const float* enc_Wq = (const float*)d_in[12];
    const float* enc_bq = (const float*)d_in[13];
    const float* enc_Wv = (const float*)d_in[14];
    const float* enc_bv = (const float*)d_in[15];
    const float* enc_Wo = (const float*)d_in[16];
    const float* enc_bo = (const float*)d_in[17];
    const float* fc_W1  = (const float*)d_in[18];
    const float* fc_b1  = (const float*)d_in[19];
    const float* fc_W2  = (const float*)d_in[20];
    const float* fc_b2  = (const float*)d_in[21];
    float* out = (float*)d_out;

    // ---- workspace carve (floats) ----
    float* W = (float*)d_ws;
    size_t o = 0;
    float* gdec = W + o; o += (size_t)SEQ * SEQ;                // softmaxed graphs
    float* genc = W + o; o += (size_t)SEQ * SEQ;
    float* Kb   = W + o; o += (size_t)HEADS * MROWS * KD;       // per-head K
    float* Vb   = W + o; o += (size_t)HEADS * MROWS * VD;       // per-head V
    float* Qb   = W + o; o += (size_t)HEADS * MROWS * KD;       // per-head Q (enc)
    float* ctx  = W + o; o += (size_t)MROWS * DIM;              // head-concat context
    float* tmp  = W + o; o += (size_t)MROWS * DIM;              // attn output
    float* hb   = W + o; o += (size_t)MROWS * DIM;              // h
    float* h2b  = W + o; o += (size_t)MROWS * DIM;              // h2
    float* mo   = W + o; o += (size_t)MROWS * DIM;              // mlp output
    float* att  = W + o; o += (size_t)HEADS * NB * SEQ * SEQ;   // scores (reused as MLP hidden)
    float* mlp_t = att;                                         // 4096x2048 fits in att region

    const long HB = (long)HEADS * NB;   // 32 attention batches

    // 0) softmax the graphs
    row_softmax<<<SEQ, 256, 0, stream>>>(g_dec, gdec, SEQ);
    row_softmax<<<SEQ, 256, 0, stream>>>(g_enc, genc, SEQ);

    // ---------------- decoder self-attention on y ----------------
    // K = y @ dec_Wk + bk ; V = y @ dec_Wv + bv   (batch over heads)
    launch_gemm(stream, 1, y_in, dec_Wk, dec_bk, Kb, MROWS, KD, DIM,
                DIM, 0,  KD, 1, (long)DIM * KD,
                KD, (long)MROWS * KD, 0, 1,  KD, 1.f, 0, HEADS);
    launch_gemm(stream, 1, y_in, dec_Wv, dec_bv, Vb, MROWS, VD, DIM,
                DIM, 0,  VD, 1, (long)DIM * VD,
                VD, (long)MROWS * VD, 0, 1,  VD, 1.f, 0, HEADS);
    // scores = (Q==K) @ K^T / sqrt(KD)   (batch over h*n)
    launch_gemm(stream, 0, Kb, Kb, nullptr, att, SEQ, SEQ, KD,
                KD, (long)SEQ * KD,  1, KD, (long)SEQ * KD,
                SEQ, (long)SEQ * SEQ, 0, 1,  0, 1.f / 8.f, 0, (int)HB);
    // causal top-KNN + softmax + blend with graph (in place)
    {
        dim3 g(SEQ, (unsigned)HB);
        topk_softmax_blend<<<g, 256, 0, stream>>>(att, gdec, 1, 0.5f);
    }
    // ctx = att @ V  -> scatter into [n][s][h*VD + v]
    launch_gemm(stream, 1, att, Vb, nullptr, ctx, SEQ, VD, SEQ,
                SEQ, (long)SEQ * SEQ,  VD, 1, (long)SEQ * VD,
                DIM, (long)VD, (long)SEQ * DIM, NB,  0, 1.f, 0, (int)HB);
    // out-proj + residual LN:  h = LN(ctx@Wo + bo + y)
    launch_gemm(stream, 1, ctx, dec_Wo, dec_bo, tmp, MROWS, DIM, DIM,
                DIM, 0,  DIM, 1, 0,  DIM, 0, 0, 1,  0, 1.f, 0, 1);
    add_layernorm<<<MROWS, 256, 0, stream>>>(tmp, y_in, hb);

    // ---------------- encoder-decoder attention ----------------
    launch_gemm(stream, 1, z_in, enc_Wk, enc_bk, Kb, MROWS, KD, DIM,
                DIM, 0,  KD, 1, (long)DIM * KD,
                KD, (long)MROWS * KD, 0, 1,  KD, 1.f, 0, HEADS);
    launch_gemm(stream, 1, z_in, enc_Wv, enc_bv, Vb, MROWS, VD, DIM,
                DIM, 0,  VD, 1, (long)DIM * VD,
                VD, (long)MROWS * VD, 0, 1,  VD, 1.f, 0, HEADS);
    launch_gemm(stream, 1, hb, enc_Wq, enc_bq, Qb, MROWS, KD, DIM,
                DIM, 0,  KD, 1, (long)DIM * KD,
                KD, (long)MROWS * KD, 0, 1,  KD, 1.f, 0, HEADS);
    launch_gemm(stream, 0, Qb, Kb, nullptr, att, SEQ, SEQ, KD,
                KD, (long)SEQ * KD,  1, KD, (long)SEQ * KD,
                SEQ, (long)SEQ * SEQ, 0, 1,  0, 1.f / 8.f, 0, (int)HB);
    {
        dim3 g(SEQ, (unsigned)HB);
        topk_softmax_blend<<<g, 256, 0, stream>>>(att, genc, 0, 0.5f);
    }
    launch_gemm(stream, 1, att, Vb, nullptr, ctx, SEQ, VD, SEQ,
                SEQ, (long)SEQ * SEQ,  VD, 1, (long)SEQ * VD,
                DIM, (long)VD, (long)SEQ * DIM, NB,  0, 1.f, 0, (int)HB);
    launch_gemm(stream, 1, ctx, enc_Wo, enc_bo, tmp, MROWS, DIM, DIM,
                DIM, 0,  DIM, 1, 0,  DIM, 0, 0, 1,  0, 1.f, 0, 1);
    add_layernorm<<<MROWS, 256, 0, stream>>>(tmp, hb, h2b);

    // ---------------- MLP + final LN ----------------
    launch_gemm(stream, 1, h2b, fc_W1, fc_b1, mlp_t, MROWS, FCD, DIM,
                DIM, 0,  FCD, 1, 0,  FCD, 0, 0, 1,  0, 1.f, 1, 1);
    launch_gemm(stream, 1, mlp_t, fc_W2, fc_b2, mo, MROWS, DIM, FCD,
                FCD, 0,  DIM, 1, 0,  DIM, 0, 0, 1,  0, 1.f, 0, 1);
    add_layernorm<<<MROWS, 256, 0, stream>>>(mo, h2b, out);
}
